// DGCNN_86809878987299
// MI455X (gfx1250) — matrix operand order repaired and checked by reading, not compiled, not run on hardware
//
#include <hip/hip_runtime.h>

typedef __attribute__((ext_vector_type(16))) _Float16 v16h;
typedef __attribute__((ext_vector_type(8)))  _Float16 v8h;
typedef __attribute__((ext_vector_type(8)))  float    v8f;

#define NPTS 2048
#define BATCH 16
#define KNN 20
#define SLOPE 0.2f
#define EPS 1e-5f
#define NEG_INF (-3.4e38f)

// ---------------------------------------------------------------- utilities
__global__ void zero256_kernel(float* p) { p[threadIdx.x] = 0.0f; }

__global__ void lrelu_cvt_kernel(const float* __restrict__ in, _Float16* __restrict__ out, int n) {
    for (int i = blockIdx.x * blockDim.x + threadIdx.x; i < n; i += gridDim.x * blockDim.x) {
        float v = in[i];
        v = (v >= 0.f) ? v : SLOPE * v;
        out[i] = (_Float16)v;
    }
}

// Pack a row-major f32 weight matrix W[Kd x Cc] into the WMMA-B fragment-native
// f16 layout: element (((kb*ntiles + tn)*32 + lane)*16 + j) holds
// W[kb*32 + 8*(lane>>4) + (j&7) + 16*(j>>3)][tn*16 + (lane&15)], so each lane's 16 fragment
// values for one k-step are 32 contiguous bytes (two global_load_b128).
__global__ void pack_b_kernel(const float* __restrict__ W, _Float16* __restrict__ out,
                              int Kd, int Cc) {
    int total = Kd * Cc;
    for (int i = blockIdx.x * blockDim.x + threadIdx.x; i < total; i += gridDim.x * blockDim.x) {
        int j    = i & 15;
        int lane = (i >> 4) & 31;
        int t2   = i >> 9;                 // kb*ntiles + tn
        int ntiles = Cc >> 4;
        int kb = t2 / ntiles, tn = t2 % ntiles;
        int K   = kb * 32 + 8 * (lane >> 4) + (j & 7) + 16 * (j >> 3);
        int col = tn * 16 + (lane & 15);
        out[i] = (_Float16)W[K * Cc + col];
    }
}

// ---------------------------------------------------------------- KNN top-k
__global__ void topk_kernel(const float* __restrict__ x, int* __restrict__ knn) {
    int pn = blockIdx.x;            // b*NPTS + n
    int b  = pn >> 11;
    int n  = pn & (NPTS - 1);
    int t  = threadIdx.x;           // 256 threads

    __shared__ float spts[NPTS * 3];   // 24 KB
    __shared__ float sval[NPTS];       // 8 KB
    __shared__ float sv[256];
    __shared__ int   si[256];

    for (int i = t; i < NPTS * 3; i += 256) spts[i] = x[b * NPTS * 3 + i];
    __syncthreads();

    float p0 = spts[n * 3 + 0], p1 = spts[n * 3 + 1], p2 = spts[n * 3 + 2];
    float xxn = p0 * p0 + p1 * p1 + p2 * p2;
    for (int m = t; m < NPTS; m += 256) {
        float q0 = spts[m * 3 + 0], q1 = spts[m * 3 + 1], q2 = spts[m * 3 + 2];
        float inner = p0 * q0 + p1 * q1 + p2 * q2;
        float xxm   = q0 * q0 + q1 * q1 + q2 * q2;
        sval[m] = 2.0f * inner - xxn - xxm;
    }
    __syncthreads();

    for (int it = 0; it < KNN; ++it) {
        float best = NEG_INF; int bi = 0x7fffffff;
        for (int m = t; m < NPTS; m += 256) {
            float v = sval[m];
            if (v > best) { best = v; bi = m; }   // ascending scan -> earliest index on tie
        }
        sv[t] = best; si[t] = bi;
        __syncthreads();
        for (int s = 128; s > 0; s >>= 1) {
            if (t < s) {
                if (sv[t + s] > sv[t] || (sv[t + s] == sv[t] && si[t + s] < si[t])) {
                    sv[t] = sv[t + s]; si[t] = si[t + s];
                }
            }
            __syncthreads();
        }
        if (t == 0) { knn[pn * KNN + it] = si[0]; sval[si[0]] = NEG_INF; }
        __syncthreads();
    }
}

// ---------------------------------------------------------------- edge conv
#define EPTS 8
__global__ void edge_stats_kernel(const float* __restrict__ x, const int* __restrict__ knn,
                                  const float* __restrict__ cw,
                                  float* __restrict__ sums, float* __restrict__ sqs) {
    int o = threadIdx.x;    // channel 0..63
    __shared__ float sxi[3];
    __shared__ int   sk[KNN];
    __shared__ float sxj[KNN * 3];
    float w[6];
#pragma unroll
    for (int c = 0; c < 6; ++c) w[c] = cw[o * 6 + c];
    float s = 0.f, q = 0.f;
    int base = blockIdx.x * EPTS;
    for (int p = 0; p < EPTS; ++p) {
        int pn = base + p;
        int b  = pn >> 11;
        __syncthreads();
        if (o < 3)   sxi[o] = x[pn * 3 + o];
        if (o < KNN) sk[o]  = knn[pn * KNN + o];
        __syncthreads();
        if (o < KNN * 3) {
            int k = o / 3, c = o % 3;
            sxj[o] = x[(b * NPTS + sk[k]) * 3 + c];
        }
        __syncthreads();
        for (int k = 0; k < KNN; ++k) {
            float f0 = sxj[k * 3 + 0] - sxi[0];
            float f1 = sxj[k * 3 + 1] - sxi[1];
            float f2 = sxj[k * 3 + 2] - sxi[2];
            float h = f0 * w[0] + f1 * w[1] + f2 * w[2]
                    + sxi[0] * w[3] + sxi[1] * w[4] + sxi[2] * w[5];
            s += h; q += h * h;
        }
    }
    atomicAdd(&sums[o], s);
    atomicAdd(&sqs[o], q);
}

__global__ void edge_apply_kernel(const float* __restrict__ x, const int* __restrict__ knn,
                                  const float* __restrict__ cw,
                                  const float* __restrict__ scale, const float* __restrict__ shift,
                                  _Float16* __restrict__ feat16) {
    int o  = threadIdx.x;
    int pn = blockIdx.x;
    int b  = pn >> 11;
    __shared__ float sxi[3];
    __shared__ int   sk[KNN];
    __shared__ float sxj[KNN * 3];
    if (o < 3)   sxi[o] = x[pn * 3 + o];
    if (o < KNN) sk[o]  = knn[pn * KNN + o];
    __syncthreads();
    if (o < KNN * 3) {
        int k = o / 3, c = o % 3;
        sxj[o] = x[(b * NPTS + sk[k]) * 3 + c];
    }
    __syncthreads();
    float w[6];
#pragma unroll
    for (int c = 0; c < 6; ++c) w[c] = cw[o * 6 + c];
    float sc = scale[o], sh = shift[o];
    float m = NEG_INF;
    for (int k = 0; k < KNN; ++k) {
        float f0 = sxj[k * 3 + 0] - sxi[0];
        float f1 = sxj[k * 3 + 1] - sxi[1];
        float f2 = sxj[k * 3 + 2] - sxi[2];
        float h = f0 * w[0] + f1 * w[1] + f2 * w[2]
                + sxi[0] * w[3] + sxi[1] * w[4] + sxi[2] * w[5];
        float v = h * sc + sh;
        v = (v >= 0.f) ? v : SLOPE * v;
        m = fmaxf(m, v);
    }
    feat16[pn * 64 + o] = (_Float16)m;
}

// ---------------------------------------------------------------- WMMA GEMM
// Y[R x C] = A[R x Kd](f16 row-major) @ Bp[Kd x C](f16, fragment-packed) + bias.
// One wave per 16x16 tile. Fragments come in as aligned 128-bit loads:
//   A: two b128 per k-step (the ISA A-layout is K-contiguous per lane),
//   B: two b128 per k-step from the packed layout produced by pack_b_kernel.
// Kd is a compile-time constant so the K-loop fully unrolls into a stream of
// v_wmma_f32_16x16x32_f16 ops with overlapped loads.
template <int Kd>
__global__ void gemm_f16_kernel(const _Float16* __restrict__ A, const _Float16* __restrict__ Bp,
                                const float* __restrict__ bias, float* __restrict__ Y,
                                int Rr, int Cc) {
    int wave = (blockIdx.x * blockDim.x + threadIdx.x) >> 5;
    int lane = threadIdx.x & 31;
    int ntiles = Cc >> 4;
    int total  = (Rr >> 4) * ntiles;
    if (wave >= total) return;                 // wave-uniform: EXEC stays all-1s
    int tm = wave / ntiles, tn = wave % ntiles;
    int khalf = lane >> 4;
    int mrow  = tm * 16 + (lane & 15);
    int ncol  = tn * 16 + (lane & 15);

    const v8h* __restrict__ pa = (const v8h*)(A + (size_t)mrow * Kd);          // v8h units
    const v8h* __restrict__ pb = (const v8h*)(Bp + ((size_t)tn * 32 + lane) * 16);
    const int bstride = ntiles * 64;           // v8h units between k-blocks of packed B

    __builtin_prefetch(pa, 0, 1);              // global_prefetch_b8
    __builtin_prefetch(pb, 0, 1);

    v8f acc = {};
#pragma unroll
    for (int kb = 0; kb < Kd / 32; ++kb) {
        v8h alo = pa[kb * 4 + khalf];          // K = kb*32 + 8*khalf     .. +7
        v8h ahi = pa[kb * 4 + khalf + 2];      // K = kb*32 + 16 + 8*khalf.. +7
        v8h blo = pb[kb * bstride + 0];        // packed: j = 0..7
        v8h bhi = pb[kb * bstride + 1];        // packed: j = 8..15
        v16h a = __builtin_shufflevector(alo, ahi, 0, 1, 2, 3, 4, 5, 6, 7,
                                         8, 9, 10, 11, 12, 13, 14, 15);
        v16h b = __builtin_shufflevector(blo, bhi, 0, 1, 2, 3, 4, 5, 6, 7,
                                         8, 9, 10, 11, 12, 13, 14, 15);
        acc = __builtin_amdgcn_wmma_f32_16x16x32_f16(false, a, false, b,
                                                     (short)0, acc, false, false);
    }
#pragma unroll
    for (int r = 0; r < 8; ++r) {
        int row = tm * 16 + r + 8 * khalf;     // C/D lane layout
        Y[row * Cc + ncol] = acc[r] + bias[ncol];
    }
}

// ---------------------------------------------------------------- BN pieces
__global__ void colstats_kernel(const float* __restrict__ y, int rows, int cols,
                                float* __restrict__ sums, float* __restrict__ sqs) {
    int tid = threadIdx.x;
    int col = tid % cols;
    int rg  = tid / cols;
    int rpb = blockDim.x / cols;
    float s = 0.f, q = 0.f;
    for (int r = blockIdx.x * rpb + rg; r < rows; r += gridDim.x * rpb) {
        float v = y[r * cols + col];
        s += v; q += v * v;
    }
    __shared__ float ss[256], sq2[256];
    ss[tid] = s; sq2[tid] = q;
    __syncthreads();
    if (tid < cols) {
        float ts = 0.f, tq = 0.f;
        for (int g = 0; g < rpb; ++g) { ts += ss[g * cols + col]; tq += sq2[g * cols + col]; }
        atomicAdd(&sums[col], ts);
        atomicAdd(&sqs[col], tq);
    }
}

__global__ void finalize_kernel(const float* __restrict__ sums, const float* __restrict__ sqs,
                                const float* __restrict__ g, const float* __restrict__ bt,
                                float* __restrict__ scale, float* __restrict__ shift,
                                float invcount) {
    int c = threadIdx.x;
    float mean = sums[c] * invcount;
    float var  = sqs[c] * invcount - mean * mean;
    float s    = g[c] * rsqrtf(var + EPS);
    scale[c] = s;
    shift[c] = bt[c] - mean * s;
}

__global__ void apply_bn_kernel(const float* __restrict__ y,
                                const float* __restrict__ scale, const float* __restrict__ shift,
                                _Float16* __restrict__ out, int n, int cols) {
    for (int i = blockIdx.x * blockDim.x + threadIdx.x; i < n; i += gridDim.x * blockDim.x) {
        int c = i % cols;
        float v = y[i] * scale[c] + shift[c];
        v = (v >= 0.f) ? v : SLOPE * v;
        out[i] = (_Float16)v;
    }
}

// ---------------------------------------------------------------- global max pool
__global__ void pool_kernel(const _Float16* __restrict__ a, _Float16* __restrict__ p) {
    int b = blockIdx.x, c = threadIdx.x;   // 16 x 128
    float m = NEG_INF;
    for (int n = 0; n < NPTS; ++n)
        m = fmaxf(m, (float)a[(b * NPTS + n) * 128 + c]);
    p[b * 128 + c] = (_Float16)m;
}

// ---------------------------------------------------------------- launch
extern "C" void kernel_launch(void* const* d_in, const int* in_sizes, int n_in,
                              void* d_out, int out_size, void* d_ws, size_t ws_size,
                              hipStream_t stream) {
    (void)in_sizes; (void)n_in; (void)out_size; (void)ws_size;
    const float* x      = (const float*)d_in[0];
    const float* conv_w = (const float*)d_in[1];
    const float* conv_g = (const float*)d_in[2];
    const float* conv_b = (const float*)d_in[3];
    const float* w1 = (const float*)d_in[4];  const float* b1  = (const float*)d_in[5];
    const float* g1 = (const float*)d_in[6];  const float* bt1 = (const float*)d_in[7];
    const float* w2 = (const float*)d_in[8];  const float* b2  = (const float*)d_in[9];
    const float* g2 = (const float*)d_in[10]; const float* bt2 = (const float*)d_in[11];
    const float* w3 = (const float*)d_in[12]; const float* b3  = (const float*)d_in[13];
    const float* g3 = (const float*)d_in[14]; const float* bt3 = (const float*)d_in[15];
    const float* w4 = (const float*)d_in[16]; const float* b4  = (const float*)d_in[17];
    const float* w5 = (const float*)d_in[18]; const float* b5  = (const float*)d_in[19];
    float* out = (float*)d_out;

    const int ROWS = BATCH * NPTS;           // 32768

    char* ws = (char*)d_ws;
    size_t off = 0;
    auto take = [&](size_t bytes) -> char* {
        char* p = ws + off;
        off = (off + bytes + 255) & ~(size_t)255;
        return p;
    };
    int*      knn    = (int*)take((size_t)ROWS * KNN * 4);          // 2.6 MB
    _Float16* feat16 = (_Float16*)take((size_t)ROWS * 64 * 2);      // 4.2 MB
    float*    y1     = (float*)take((size_t)ROWS * 64 * 4);         // 8.4 MB (reused as act3)
    _Float16* act1   = (_Float16*)take((size_t)ROWS * 64 * 2);      // 4.2 MB
    float*    y2     = (float*)take((size_t)ROWS * 128 * 4);        // 16.8 MB (reused as y3)
    _Float16* act2   = (_Float16*)take((size_t)ROWS * 128 * 2);     // 8.4 MB
    _Float16* pooled = (_Float16*)take(BATCH * 128 * 2);
    float*    y4     = (float*)take(BATCH * 512 * 4);
    _Float16* act4   = (_Float16*)take(BATCH * 512 * 2);
    _Float16* w1h    = (_Float16*)take(64 * 64 * 2);     // fragment-packed weights
    _Float16* w2h    = (_Float16*)take(64 * 128 * 2);
    _Float16* w3h    = (_Float16*)take(128 * 128 * 2);
    _Float16* w4h    = (_Float16*)take(128 * 512 * 2);
    _Float16* w5h    = (_Float16*)take(512 * 256 * 2);
    float*    sums   = (float*)take(512);    // contiguous with sqs (zeroed together)
    float*    sqs    = (float*)take(512);
    float*    scale  = (float*)take(512);
    float*    shift  = (float*)take(512);
    float*    y3     = y2;                    // y2 dead before gemm3 writes
    _Float16* act3   = (_Float16*)y1;         // y1 dead before apply3 writes

    // weights -> fragment-packed f16, once per call
    pack_b_kernel<<<16, 256, 0, stream>>>(w1, w1h, 64, 64);
    pack_b_kernel<<<32, 256, 0, stream>>>(w2, w2h, 64, 128);
    pack_b_kernel<<<64, 256, 0, stream>>>(w3, w3h, 128, 128);
    pack_b_kernel<<<256, 256, 0, stream>>>(w4, w4h, 128, 512);
    pack_b_kernel<<<512, 256, 0, stream>>>(w5, w5h, 512, 256);

    // KNN
    topk_kernel<<<ROWS, 256, 0, stream>>>(x, knn);

    // edge conv: stats -> affine -> apply+maxk
    zero256_kernel<<<1, 256, 0, stream>>>(sums);
    edge_stats_kernel<<<ROWS / EPTS, 64, 0, stream>>>(x, knn, conv_w, sums, sqs);
    finalize_kernel<<<1, 64, 0, stream>>>(sums, sqs, conv_g, conv_b, scale, shift,
                                          1.0f / (float)(ROWS * KNN));
    edge_apply_kernel<<<ROWS, 64, 0, stream>>>(x, knn, conv_w, scale, shift, feat16);

    // layer 1: 32768x64 @ 64x64
    gemm_f16_kernel<64><<<(ROWS / 16) * (64 / 16) / 8, 256, 0, stream>>>(feat16, w1h, b1, y1, ROWS, 64);
    zero256_kernel<<<1, 256, 0, stream>>>(sums);
    colstats_kernel<<<256, 256, 0, stream>>>(y1, ROWS, 64, sums, sqs);
    finalize_kernel<<<1, 64, 0, stream>>>(sums, sqs, g1, bt1, scale, shift, 1.0f / (float)ROWS);
    apply_bn_kernel<<<2048, 256, 0, stream>>>(y1, scale, shift, act1, ROWS * 64, 64);

    // layer 2: 32768x64 @ 64x128
    gemm_f16_kernel<64><<<(ROWS / 16) * (128 / 16) / 8, 256, 0, stream>>>(act1, w2h, b2, y2, ROWS, 128);
    zero256_kernel<<<1, 256, 0, stream>>>(sums);
    colstats_kernel<<<256, 256, 0, stream>>>(y2, ROWS, 128, sums, sqs);
    finalize_kernel<<<1, 128, 0, stream>>>(sums, sqs, g2, bt2, scale, shift, 1.0f / (float)ROWS);
    apply_bn_kernel<<<4096, 256, 0, stream>>>(y2, scale, shift, act2, ROWS * 128, 128);

    // layer 3: 32768x128 @ 128x128
    gemm_f16_kernel<128><<<(ROWS / 16) * (128 / 16) / 8, 256, 0, stream>>>(act2, w3h, b3, y3, ROWS, 128);
    zero256_kernel<<<1, 256, 0, stream>>>(sums);
    colstats_kernel<<<256, 256, 0, stream>>>(y3, ROWS, 128, sums, sqs);
    finalize_kernel<<<1, 128, 0, stream>>>(sums, sqs, g3, bt3, scale, shift, 1.0f / (float)ROWS);
    apply_bn_kernel<<<4096, 256, 0, stream>>>(y3, scale, shift, act3, ROWS * 128, 128);

    // global max pool over N
    pool_kernel<<<BATCH, 128, 0, stream>>>(act3, pooled);

    // FC 128->512 (+lrelu), FC 512->256 -> out
    gemm_f16_kernel<128><<<(16 / 16) * (512 / 16) / 8, 256, 0, stream>>>(pooled, w4h, b4, y4, 16, 512);
    lrelu_cvt_kernel<<<32, 256, 0, stream>>>(y4, act4, 16 * 512);
    gemm_f16_kernel<512><<<(16 / 16) * (256 / 16) / 8, 256, 0, stream>>>(act4, w5h, b5, out, 16, 256);
}
